// Qwen35ForANE_59322088293088
// MI455X (gfx1250) — matrix operand order repaired and checked by reading, not compiled, not run on hardware
//
#include <hip/hip_runtime.h>
#include <hip/hip_bf16.h>

// ---------------- problem constants ----------------
#define C_  2048
#define H_  16
#define HK_ 4
#define D_  128
#define S_  1024

typedef __attribute__((ext_vector_type(16))) __bf16 bf16x16;
typedef __attribute__((ext_vector_type(8)))  __bf16 bf16x8;
typedef __attribute__((ext_vector_type(8)))  float  floatx8;
typedef __attribute__((ext_vector_type(4)))  unsigned int uint32x4;
typedef __attribute__((ext_vector_type(8)))  int          int32x8;
typedef __attribute__((ext_vector_type(4)))  int          int32x4;

#define RFL(x) ((unsigned int)__builtin_amdgcn_readfirstlane((int)(x)))

// =====================================================================
// Tensor Data Mover: 2D tile (global -> LDS). All units are 8-byte
// (data_size=3) elements. One instruction moves the whole tile; tracked
// by TENSORcnt. D# packing per CDNA5 ISA 8.3/8.4 (2D: groups 2,3 = 0).
// Toolchain has the 6-arg builtin: (g0, g1, g2, g3, g4, cpol).
// =====================================================================
__device__ __forceinline__ void tdm_load_2d(unsigned int lds_off, const void* gptr,
                                            unsigned int tensor_d0,  // row len (units)
                                            unsigned int tensor_d1,  // #rows
                                            unsigned int tile_d0,    // tile row len (units)
                                            unsigned int tile_d1,    // tile #rows
                                            unsigned int stride0) {  // row pitch (units)
  const unsigned long long ga = (unsigned long long)gptr;
  uint32x4 g0;
  g0[0] = RFL(1u);                                   // count=1, user mode
  g0[1] = RFL(lds_off);                              // lds_addr (bytes)
  g0[2] = RFL((unsigned int)ga);                     // global_addr[31:0]
  g0[3] = RFL(((unsigned int)(ga >> 32) & 0x1FFFFFFu) | (2u << 30));  // ga[56:32] | type=2
  int32x8 g1;
  g1[0] = (int)RFL(3u << 16);                                          // data_size=8B
  g1[1] = (int)RFL((tensor_d0 & 0xFFFFu) << 16);                       // td0[15:0]
  g1[2] = (int)RFL((tensor_d0 >> 16) | ((tensor_d1 & 0xFFFFu) << 16)); // td0[31:16]|td1[15:0]
  g1[3] = (int)RFL((tensor_d1 >> 16) | (tile_d0 << 16));               // td1[31:16]|tile_d0
  g1[4] = (int)RFL(tile_d1);                                           // tile_d1, tile_d2=0
  g1[5] = (int)RFL(stride0);                                           // stride0[31:0]
  g1[6] = 0;                                                           // stride0[47:32], stride1[15:0]
  g1[7] = 0;
  const int32x4 z4 = {0, 0, 0, 0};
  const int32x8 z8 = {0, 0, 0, 0, 0, 0, 0, 0};
  __builtin_amdgcn_tensor_load_to_lds(g0, g1, z4, z4, z8, 0);
}

template <short N>
__device__ __forceinline__ void tdm_wait() {
  __builtin_amdgcn_s_wait_tensorcnt(N);
}

__device__ __forceinline__ unsigned int lds_off_of(const void* p) {
  return (unsigned int)(size_t)p;  // LDS aperture: low 32 bits == LDS byte offset
}

// =====================================================================
// GEMM: out[M][N] = A[M][K] @ B[N][K]^T, A/B bf16 row-major, out fp32.
// Block tile 128x128, K-step 64, double-buffered TDM tiles.
// 8 waves; wave w owns rows [16w,16w+16) x 128 cols (8 C fragments).
// =====================================================================
__global__ __launch_bounds__(256)
void gemm_wmma_tdm(const __bf16* __restrict__ A, const __bf16* __restrict__ B,
                   float* __restrict__ out, int M, int N, int K) {
  __shared__ alignas(128) __bf16 As[2][128][64];   // 2 x 16 KB
  __shared__ alignas(128) __bf16 Bs[2][128][64];   // 2 x 16 KB
  const int tid = threadIdx.x;
  const int wave = tid >> 5, lane = tid & 31;
  const int hi = lane >> 4, ln = lane & 15;
  const int m0 = blockIdx.y * 128, n0 = blockIdx.x * 128;
  const int NK = K >> 6;
  const unsigned int rowu = (unsigned int)(K >> 2);  // row length/pitch in 8B units

  floatx8 c[8] = {};

  if (wave == 0) {
    tdm_load_2d(lds_off_of(&As[0][0][0]), A + (size_t)m0 * K, rowu, M, 16, 128, rowu);
    tdm_load_2d(lds_off_of(&Bs[0][0][0]), B + (size_t)n0 * K, rowu, N, 16, 128, rowu);
  }

  for (int s = 0; s < NK; ++s) {
    const int cur = s & 1, nxt = cur ^ 1;
    __syncthreads();  // everyone done reading buf[nxt] (stage s-1)
    if (wave == 0) {
      if (s + 1 < NK) {
        const size_t koff = (size_t)(s + 1) << 6;
        tdm_load_2d(lds_off_of(&As[nxt][0][0]), A + (size_t)m0 * K + koff, rowu, M, 16, 128, rowu);
        tdm_load_2d(lds_off_of(&Bs[nxt][0][0]), B + (size_t)n0 * K + koff, rowu, N, 16, 128, rowu);
        tdm_wait<2>();  // stage-s pair complete (in-order), s+1 pair may fly
      } else {
        tdm_wait<0>();
      }
    }
    __syncthreads();  // stage-s tiles visible to all waves

    // A fragments (2 k-chunks): lanes 0-15 k{0..15}, lanes 16-31 k{16..31}
    bf16x16 a[2];
    {
      const __bf16* arow = &As[cur][wave * 16 + ln][0];
#pragma unroll
      for (int kc = 0; kc < 2; ++kc) {
        bf16x8 lo = *(const bf16x8*)(arow + kc * 32 + 16 * hi);
        bf16x8 h8 = *(const bf16x8*)(arow + kc * 32 + 16 * hi + 8);
#pragma unroll
        for (int i = 0; i < 8; ++i) { a[kc][i] = lo[i]; a[kc][i + 8] = h8[i]; }
      }
    }
#pragma unroll
    for (int cf = 0; cf < 8; ++cf) {
#pragma unroll
      for (int kc = 0; kc < 2; ++kc) {
        bf16x16 b = *(const bf16x16*)(&Bs[cur][16 * cf + ln][kc * 32 + 16 * hi]);
        c[cf] = __builtin_amdgcn_wmma_f32_16x16x32_bf16(false, a[kc], false, b,
                                                        (short)0, c[cf], false, false);
      }
    }
  }

#pragma unroll
  for (int cf = 0; cf < 8; ++cf)
#pragma unroll
    for (int r = 0; r < 8; ++r)
      out[(size_t)(m0 + wave * 16 + 8 * hi + r) * N + n0 + 16 * cf + ln] = c[cf][r];
}

// =====================================================================
// fp32 -> bf16, 8 elements/thread (vectorized)
// =====================================================================
__global__ __launch_bounds__(256)
void cvt_bf16_kernel(const float* __restrict__ in, __bf16* __restrict__ outp) {
  const size_t i = ((size_t)blockIdx.x * 256 + threadIdx.x) * 8;
  const float4 x = *(const float4*)(in + i);
  const float4 y = *(const float4*)(in + i + 4);
  bf16x8 v;
  v[0] = (__bf16)x.x; v[1] = (__bf16)x.y; v[2] = (__bf16)x.z; v[3] = (__bf16)x.w;
  v[4] = (__bf16)y.x; v[5] = (__bf16)y.y; v[6] = (__bf16)y.z; v[7] = (__bf16)y.w;
  *(bf16x8*)(outp + i) = v;
}

// =====================================================================
// transpose + convert: in fp32 [R][Ccols] -> out bf16 [Ccols][R]
// =====================================================================
__global__ __launch_bounds__(256)
void transpose_cvt_kernel(const float* __restrict__ in, __bf16* __restrict__ outp,
                          int R, int Ccols) {
  __shared__ __bf16 t[64][65];
  const int r0 = blockIdx.y * 64, c0 = blockIdx.x * 64;
  const int tid = threadIdx.x;
#pragma unroll
  for (int i = 0; i < 4; ++i) {
    const int j = tid + 256 * i;           // 0..1023
    const int r = j >> 4, c4 = (j & 15) * 4;
    const float4 f = *(const float4*)(in + (size_t)(r0 + r) * Ccols + c0 + c4);
    t[c4 + 0][r] = (__bf16)f.x; t[c4 + 1][r] = (__bf16)f.y;
    t[c4 + 2][r] = (__bf16)f.z; t[c4 + 3][r] = (__bf16)f.w;
  }
  __syncthreads();
#pragma unroll
  for (int i = 0; i < 4; ++i) {
    const int j = tid + 256 * i;
    const int cc = j >> 4, r4 = (j & 15) * 4;
    __bf16* o = outp + (size_t)(c0 + cc) * R + r0 + r4;
    o[0] = t[cc][r4]; o[1] = t[cc][r4 + 1]; o[2] = t[cc][r4 + 2]; o[3] = t[cc][r4 + 3];
  }
}

// =====================================================================
// RMSNorm + RoPE: src rows [h*head_stride_rows + d][S] -> bf16 [h][s][d]
// =====================================================================
__global__ __launch_bounds__(128)
void rmsrope_kernel(const float* __restrict__ src, const float* __restrict__ norm_w,
                    const float* __restrict__ cosc, const float* __restrict__ sinc,
                    const int* __restrict__ pos_ids,
                    __bf16* __restrict__ dst, int head_stride_rows) {
  const int s = blockIdx.x * 128 + threadIdx.x;
  const int h = blockIdx.y;
  const float* base = src + (size_t)h * head_stride_rows * S_ + s;

  float ss = 0.f;
#pragma unroll 8
  for (int d = 0; d < 128; ++d) { const float v = base[(size_t)d * S_]; ss += v * v; }
  const float r = rsqrtf(ss * (1.f / 128.f) + 1e-6f);

  const int pos = pos_ids[s];
  __bf16* o = dst + ((size_t)h * S_ + s) * 128;

  float rot[32];
#pragma unroll
  for (int d = 0; d < 32; ++d) rot[d] = base[(size_t)d * S_] * r * (1.f + norm_w[d]);
#pragma unroll
  for (int d = 0; d < 32; ++d) {
    const float cs = cosc[pos * 32 + d], sn = sinc[pos * 32 + d];
    const float other = (d < 16) ? -rot[d + 16] : rot[d - 16];
    o[d] = (__bf16)(rot[d] * cs + other * sn);
  }
  for (int d = 32; d < 128; ++d)
    o[d] = (__bf16)(base[(size_t)d * S_] * r * (1.f + norm_w[d]));
}

// =====================================================================
// Flash attention + sigmoid gate. Block = 1 head x 128 queries, 8 waves.
// K/V tiles arrive via TDM; Q fragments live in registers.
// Output written bf16 as att_t[s][h*D+d]  (== B-operand layout for o_proj).
// =====================================================================
__global__ __launch_bounds__(256)
void attention_kernel(const __bf16* __restrict__ qb, const __bf16* __restrict__ kbf,
                      const __bf16* __restrict__ vbt, const float* __restrict__ q_raw,
                      __bf16* __restrict__ att_t) {
  const int h = blockIdx.y, qblk = blockIdx.x;
  __shared__ alignas(128) __bf16 Ks[64][128];     // [key][dim]   16 KB
  __shared__ alignas(128) __bf16 Vt[128][64];     // [dim][key]   16 KB
  __shared__ alignas(128) __bf16 Ps[8][16][64];   // per-wave P   16 KB

  const int tid = threadIdx.x, wave = tid >> 5, lane = tid & 31;
  const int hi = lane >> 4, ln = lane & 15;

  // Q fragments (4 k-chunks of 32) straight from global in A-layout
  bf16x16 qf[4];
  {
    const __bf16* qrow = qb + ((size_t)h * S_ + qblk * 128 + wave * 16 + ln) * 128;
#pragma unroll
    for (int kc = 0; kc < 4; ++kc) {
      bf16x8 lo = *(const bf16x8*)(qrow + kc * 32 + 16 * hi);
      bf16x8 h8 = *(const bf16x8*)(qrow + kc * 32 + 16 * hi + 8);
#pragma unroll
      for (int i = 0; i < 8; ++i) { qf[kc][i] = lo[i]; qf[kc][i + 8] = h8[i]; }
    }
  }

  float m_r[8], l_r[8];
  floatx8 acc[8];
#pragma unroll
  for (int r = 0; r < 8; ++r) { m_r[r] = -3.0e38f; l_r[r] = 0.f; }
#pragma unroll
  for (int d = 0; d < 8; ++d) acc[d] = (floatx8){};

  const int q_base = qblk * 128 + wave * 16 + 8 * hi;
  const int kb_end = (qblk + 1) * 128;  // causal: skip strictly-future blocks

  for (int kb0 = 0; kb0 < kb_end; kb0 += 64) {
    __syncthreads();  // previous tile fully consumed
    if (wave == 0) {
      // K tile: 64 keys x 32 units, contiguous rows
      tdm_load_2d(lds_off_of(&Ks[0][0]), kbf + ((size_t)h * S_ + kb0) * 128,
                  32, 64, 32, 64, 32);
      // V^T tile: 128 dim-rows x 16 units, pitch S*2B = 256 units
      tdm_load_2d(lds_off_of(&Vt[0][0]), vbt + ((size_t)h * 128) * S_ + kb0,
                  256, 128, 16, 128, 256);
      tdm_wait<0>();
    }
    __syncthreads();

    // S = Q K^T (16 WMMAs/wave)
    floatx8 sc[4];
#pragma unroll
    for (int cf = 0; cf < 4; ++cf) {
      sc[cf] = (floatx8){};
#pragma unroll
      for (int kc = 0; kc < 4; ++kc) {
        bf16x16 b = *(const bf16x16*)(&Ks[16 * cf + ln][kc * 32 + 16 * hi]);
        sc[cf] = __builtin_amdgcn_wmma_f32_16x16x32_bf16(false, qf[kc], false, b,
                                                         (short)0, sc[cf], false, false);
      }
    }

    const float scale = 0.08838834764831845f;  // 1/sqrt(128)
#pragma unroll
    for (int cf = 0; cf < 4; ++cf) {
      const int kk = kb0 + 16 * cf + ln;
#pragma unroll
      for (int r = 0; r < 8; ++r)
        sc[cf][r] = (kk <= q_base + r) ? sc[cf][r] * scale : -1.0e9f;
    }

    // online softmax (rows live in 16-lane halves)
#pragma unroll
    for (int r = 0; r < 8; ++r) {
      float mx = fmaxf(fmaxf(sc[0][r], sc[1][r]), fmaxf(sc[2][r], sc[3][r]));
#pragma unroll
      for (int off = 1; off < 16; off <<= 1) mx = fmaxf(mx, __shfl_xor(mx, off, 32));
      const float mnew = fmaxf(m_r[r], mx);
      const float alpha = __expf(m_r[r] - mnew);
      float rs = 0.f;
#pragma unroll
      for (int cf = 0; cf < 4; ++cf) {
        const float p = __expf(sc[cf][r] - mnew);
        sc[cf][r] = p; rs += p;
      }
#pragma unroll
      for (int off = 1; off < 16; off <<= 1) rs += __shfl_xor(rs, off, 32);
      l_r[r] = l_r[r] * alpha + rs;
      m_r[r] = mnew;
#pragma unroll
      for (int d = 0; d < 8; ++d) acc[d][r] *= alpha;
    }

    // P (C-layout) -> per-wave LDS (intra-wave LDS in-order, no barrier)
#pragma unroll
    for (int cf = 0; cf < 4; ++cf)
#pragma unroll
      for (int r = 0; r < 8; ++r)
        Ps[wave][8 * hi + r][16 * cf + ln] = (__bf16)sc[cf][r];

    // O += P V  (16 WMMAs/wave)
#pragma unroll
    for (int kc2 = 0; kc2 < 2; ++kc2) {
      bf16x16 ap;
      {
        const __bf16* prow = &Ps[wave][ln][kc2 * 32];
        bf16x8 lo = *(const bf16x8*)(prow + 16 * hi);
        bf16x8 h8 = *(const bf16x8*)(prow + 16 * hi + 8);
#pragma unroll
        for (int i = 0; i < 8; ++i) { ap[i] = lo[i]; ap[i + 8] = h8[i]; }
      }
#pragma unroll
      for (int df = 0; df < 8; ++df) {
        bf16x16 b = *(const bf16x16*)(&Vt[16 * df + ln][kc2 * 32 + 16 * hi]);
        acc[df] = __builtin_amdgcn_wmma_f32_16x16x32_bf16(false, ap, false, b,
                                                          (short)0, acc[df], false, false);
      }
    }
  }

  // epilogue: normalize, sigmoid gate, write bf16 att_t[s][h*128+d]
#pragma unroll
  for (int df = 0; df < 8; ++df) {
    const int d = 16 * df + ln;
#pragma unroll
    for (int r = 0; r < 8; ++r) {
      const int s = q_base + r;
      const float o = acc[df][r] / l_r[r];
      const float g = q_raw[((size_t)(h * 256 + 128 + d)) * S_ + s];
      att_t[(size_t)s * 2048 + h * 128 + d] = (__bf16)(o * (1.f / (1.f + __expf(-g))));
    }
  }
}

// =====================================================================
extern "C" void kernel_launch(void* const* d_in, const int* in_sizes, int n_in,
                              void* d_out, int out_size, void* d_ws, size_t ws_size,
                              hipStream_t stream) {
  (void)in_sizes; (void)n_in; (void)out_size; (void)ws_size;
  const float* hidden = (const float*)d_in[0];   // [2048][1024]
  const float* q_w    = (const float*)d_in[1];   // [4096][2048]
  const float* k_w    = (const float*)d_in[2];   // [512][2048]
  const float* v_w    = (const float*)d_in[3];   // [512][2048]
  const float* o_w    = (const float*)d_in[4];   // [2048][2048]
  const float* q_nw   = (const float*)d_in[5];
  const float* k_nw   = (const float*)d_in[6];
  const int*   pos    = (const int*)d_in[9];
  const float* cosc   = (const float*)d_in[10];
  const float* sinc   = (const float*)d_in[11];
  float* out = (float*)d_out;                    // [2048][1024]

  // workspace layout (~64 MB)
  float*  q_raw = (float*)d_ws;                          // 4096*1024 f32
  float*  k_raw = q_raw + (size_t)4096 * 1024;           //  512*1024 f32
  float*  v_raw = k_raw + (size_t)512 * 1024;            //  512*1024 f32
  __bf16* qbn   = (__bf16*)(v_raw + (size_t)512 * 1024); // H*S*D
  __bf16* kbn   = qbn + (size_t)H_ * S_ * D_;            // HK*S*D
  __bf16* vbt   = kbn + (size_t)HK_ * S_ * D_;           // HK*D*S
  __bf16* att_t = vbt + (size_t)HK_ * D_ * S_;           // [S][H*D]
  __bf16* xt    = att_t + (size_t)S_ * H_ * D_;          // [S][C] hidden^T
  __bf16* wq_bf = xt + (size_t)S_ * C_;                  // [4096][2048]
  __bf16* wk_bf = wq_bf + (size_t)4096 * C_;             // [512][2048]
  __bf16* wv_bf = wk_bf + (size_t)512 * C_;              // [512][2048]
  __bf16* wo_bf = wv_bf + (size_t)512 * C_;              // [2048][2048]

  const dim3 blk(256);
  // --- precision staging: weights -> bf16, hidden -> bf16 transposed [S][C] ---
  cvt_bf16_kernel<<<dim3((4096 * C_) / 2048), blk, 0, stream>>>(q_w, wq_bf);
  cvt_bf16_kernel<<<dim3((512  * C_) / 2048), blk, 0, stream>>>(k_w, wk_bf);
  cvt_bf16_kernel<<<dim3((512  * C_) / 2048), blk, 0, stream>>>(v_w, wv_bf);
  cvt_bf16_kernel<<<dim3((C_   * C_) / 2048), blk, 0, stream>>>(o_w, wo_bf);
  transpose_cvt_kernel<<<dim3(S_ / 64, C_ / 64), blk, 0, stream>>>(hidden, xt, C_, S_);
  // --- projections (TDM + WMMA) ---
  gemm_wmma_tdm<<<dim3(S_ / 128, 4096 / 128), blk, 0, stream>>>(wq_bf, xt, q_raw, 4096, S_, C_);
  gemm_wmma_tdm<<<dim3(S_ / 128,  512 / 128), blk, 0, stream>>>(wk_bf, xt, k_raw,  512, S_, C_);
  gemm_wmma_tdm<<<dim3(S_ / 128,  512 / 128), blk, 0, stream>>>(wv_bf, xt, v_raw,  512, S_, C_);
  // --- rmsnorm + rope ---
  rmsrope_kernel<<<dim3(S_ / 128, H_),  dim3(128), 0, stream>>>(q_raw, q_nw, cosc, sinc, pos, qbn, 2 * D_);
  rmsrope_kernel<<<dim3(S_ / 128, HK_), dim3(128), 0, stream>>>(k_raw, k_nw, cosc, sinc, pos, kbn, D_);
  // V -> bf16 ([h][d][s] already == V^T rows)
  cvt_bf16_kernel<<<dim3((HK_ * D_ * S_) / 2048), blk, 0, stream>>>(v_raw, vbt);
  // --- attention + gate (writes bf16 [S][H*D]) ---
  attention_kernel<<<dim3(S_ / 128, H_), blk, 0, stream>>>(qbn, kbn, vbt, q_raw, att_t);
  // --- output projection ---
  gemm_wmma_tdm<<<dim3(S_ / 128, C_ / 128), blk, 0, stream>>>(wo_bf, att_t, out, C_, S_, C_);
}
